// TransformerBlock_82824149336101
// MI455X (gfx1250) — matrix operand order repaired and checked by reading, not compiled, not run on hardware
//
#include <hip/hip_runtime.h>
#include <cstddef>
#include <cstdint>

#define B_   4
#define S_   2048
#define D_   2048
#define H_   16
#define DH_  128
#define DFF_ 8192
#define BS_  (B_*S_)

typedef __attribute__((ext_vector_type(16))) __bf16         v16bf;
typedef __attribute__((ext_vector_type(8)))  float          v8f;
typedef __attribute__((ext_vector_type(16))) unsigned short v16u;
typedef __attribute__((ext_vector_type(8)))  unsigned short v8u;

__device__ __forceinline__ unsigned short f32_bf16(float f) {
  union { float f; unsigned u; } x; x.f = f;
  unsigned r = x.u + 0x7fffu + ((x.u >> 16) & 1u);
  return (unsigned short)(r >> 16);
}
__device__ __forceinline__ float bf16_f32(unsigned short h) {
  union { unsigned u; float f; } x; x.u = ((unsigned)h) << 16;
  return x.f;
}
__device__ __forceinline__ v8f wmma_bf16(v16bf a, v16bf b, v8f c) {
  return __builtin_amdgcn_wmma_f32_16x16x32_bf16(false, a, false, b, (short)0, c, false, false);
}

// Async 16B copy global -> LDS (CDNA5 GLOBAL_LOAD_ASYNC_TO_LDS_B128, tracked by ASYNCcnt).
// LDS byte address = low 32 bits of the generic pointer.
__device__ __forceinline__ void async_ld16(void* lds, const void* g) {
  unsigned off = (unsigned)(uintptr_t)lds;
  asm volatile("global_load_async_to_lds_b128 %0, %1, off"
               :: "v"(off), "v"(g)
               : "memory");
}
__device__ __forceinline__ void wait_async0() {
  asm volatile("s_wait_asynccnt 0" ::: "memory");
}

// A-operand fragment (16x32 bf16): lane&15 = M row; per lane-half K chunks {8*half..8*half+7, 16+8*half..}
__device__ __forceinline__ v16bf frag_a(const unsigned short* base, int ld, int lane, int k0) {
  const int row = lane & 15, half = lane >> 4;
  const unsigned short* p = base + (size_t)row * ld + k0 + 8 * half;
  v8u lo = *(const v8u*)p;
  v8u hi = *(const v8u*)(p + 16);
  v16u t;
#pragma unroll
  for (int j = 0; j < 8; j++) { t[j] = lo[j]; t[j + 8] = hi[j]; }
  return __builtin_bit_cast(v16bf, t);
}
// B-operand fragment (32x16 bf16, fed as W[n][k] row-major): lane&15 = N; per lane-half K chunks {8*half..8*half+7, 16+8*half..}
__device__ __forceinline__ v16bf frag_b(const unsigned short* base, int ld, int lane, int k0) {
  const int row = lane & 15, half = lane >> 4;
  const unsigned short* p = base + (size_t)row * ld + k0 + 8 * half;
  v8u lo = *(const v8u*)p;
  v8u hi = *(const v8u*)(p + 16);
  v16u t;
#pragma unroll
  for (int j = 0; j < 8; j++) { t[j] = lo[j]; t[j + 8] = hi[j]; }
  return __builtin_bit_cast(v16bf, t);
}

// ---------------- fp32 -> bf16 convert ----------------
__global__ void cvt_bf16_kernel(const float* __restrict__ in, unsigned short* __restrict__ out, size_t n) {
  size_t i = (size_t)blockIdx.x * blockDim.x + threadIdx.x;
  size_t stride = (size_t)gridDim.x * blockDim.x;
  for (; i < n; i += stride) out[i] = f32_bf16(in[i]);
}

// ---------------- RMSNorm (row per block) ----------------
__global__ __launch_bounds__(256) void rmsnorm_kernel(const float* __restrict__ x,
                                                      const float* __restrict__ g,
                                                      unsigned short* __restrict__ o) {
  const int row = blockIdx.x;
  const float* xr = x + (size_t)row * D_;
  float ss = 0.f;
  for (int c = threadIdx.x; c < D_; c += 256) { float v = xr[c]; ss += v * v; }
  __shared__ float red[256];
  red[threadIdx.x] = ss;
  __syncthreads();
  for (int s = 128; s > 0; s >>= 1) {
    if (threadIdx.x < s) red[threadIdx.x] += red[threadIdx.x + s];
    __syncthreads();
  }
  float sc = rsqrtf(red[0] * (1.0f / D_) + 1e-5f);
  unsigned short* orow = o + (size_t)row * D_;
  for (int c = threadIdx.x; c < D_; c += 256) orow[c] = f32_bf16(xr[c] * sc * g[c]);
}

// ---------------- tiled bf16 WMMA GEMM:  C[m,n] = sum_k A[m,k]*W[n,k] ----------------
// Double-buffered LDS staging via global_load_async_to_lds_b128 (ASYNCcnt pipeline).
// MODE 0: outF = resid + acc (fp32 [M,N]);  MODE 1: bf16 [b,h,s,dh];  MODE 2: bf16 [b,h,dh,s]
template <int MODE>
__global__ __launch_bounds__(256, 1) void gemm_ws_kernel(const unsigned short* __restrict__ A,
                                                         const unsigned short* __restrict__ W,
                                                         const float* __restrict__ resid,
                                                         float* __restrict__ outF,
                                                         unsigned short* __restrict__ outH,
                                                         int M, int N, int K) {
  __shared__ __align__(16) unsigned short As[2][128 * 32];
  __shared__ __align__(16) unsigned short Ws[2][64 * 32];
  const int tid = threadIdx.x;
  const int wave = tid >> 5, lane = tid & 31;
  const int lane15 = lane & 15, half = lane >> 4;
  const int n0 = blockIdx.x * 64;
  const int m0 = blockIdx.y * 128;

  v8f acc[4];
#pragma unroll
  for (int nt = 0; nt < 4; nt++)
#pragma unroll
    for (int i = 0; i < 8; i++) acc[nt][i] = 0.f;

  auto stage = [&](int buf, int k0) {
#pragma unroll
    for (int i = 0; i < 2; i++) {
      int id = tid + 256 * i;
      int row = id >> 2, cc = id & 3;
      async_ld16(&As[buf][row * 32 + cc * 8], A + (size_t)(m0 + row) * K + k0 + cc * 8);
    }
    int row = tid >> 2, cc = tid & 3;
    async_ld16(&Ws[buf][row * 32 + cc * 8], W + (size_t)(n0 + row) * K + k0 + cc * 8);
  };

  stage(0, 0);
  const int nk = K >> 5;
  for (int ki = 0; ki < nk; ki++) {
    const int cur = ki & 1;
    wait_async0();
    __syncthreads();  // tile[cur] visible to all waves
    if (ki + 1 < nk) stage(cur ^ 1, (ki + 1) * 32);  // prefetch next tile while computing
    // load all fragments into independent registers first, then chain the WMMAs,
    // so DS loads stay in flight underneath the matrix ops
    v16bf a = frag_a(&As[cur][wave * 16 * 32], 32, lane, 0);
    v16bf bfr[4];
#pragma unroll
    for (int nt = 0; nt < 4; nt++) bfr[nt] = frag_b(&Ws[cur][nt * 16 * 32], 32, lane, 0);
#pragma unroll
    for (int nt = 0; nt < 4; nt++) acc[nt] = wmma_bf16(a, bfr[nt], acc[nt]);
    __syncthreads();  // all waves done reading tile[cur] before it is overwritten
  }

#pragma unroll
  for (int nt = 0; nt < 4; nt++)
#pragma unroll
    for (int r = 0; r < 8; r++) {
      int mm = m0 + wave * 16 + r + 8 * half;
      int nn = n0 + nt * 16 + lane15;
      float v = acc[nt][r];
      if constexpr (MODE == 0) {
        size_t idx = (size_t)mm * N + nn;
        outF[idx] = resid[idx] + v;
      } else {
        int b = mm / S_, s = mm % S_;
        int h = nn / DH_, dh = nn % DH_;
        size_t idx = (MODE == 1) ? ((size_t)((b * H_ + h) * S_ + s)) * DH_ + dh
                                 : ((size_t)((b * H_ + h) * DH_ + dh)) * S_ + s;
        outH[idx] = f32_bf16(v);
      }
    }
}

// ---------------- fused SwiGLU gate GEMM: g = silu(A*W1^T) * (A*W3^T) ----------------
__global__ __launch_bounds__(256, 1) void ffn_gate_kernel(const unsigned short* __restrict__ A,
                                                          const unsigned short* __restrict__ W1,
                                                          const unsigned short* __restrict__ W3,
                                                          unsigned short* __restrict__ G,
                                                          int M, int N, int K) {
  __shared__ __align__(16) unsigned short As[2][128 * 32];
  __shared__ __align__(16) unsigned short W1s[2][64 * 32];
  __shared__ __align__(16) unsigned short W3s[2][64 * 32];
  const int tid = threadIdx.x;
  const int wave = tid >> 5, lane = tid & 31;
  const int lane15 = lane & 15, half = lane >> 4;
  const int n0 = blockIdx.x * 64;
  const int m0 = blockIdx.y * 128;

  v8f acc1[4], acc3[4];
#pragma unroll
  for (int nt = 0; nt < 4; nt++)
#pragma unroll
    for (int i = 0; i < 8; i++) { acc1[nt][i] = 0.f; acc3[nt][i] = 0.f; }

  auto stage = [&](int buf, int k0) {
#pragma unroll
    for (int i = 0; i < 2; i++) {
      int id = tid + 256 * i;
      int row = id >> 2, cc = id & 3;
      async_ld16(&As[buf][row * 32 + cc * 8], A + (size_t)(m0 + row) * K + k0 + cc * 8);
    }
    int row = tid >> 2, cc = tid & 3;
    async_ld16(&W1s[buf][row * 32 + cc * 8], W1 + (size_t)(n0 + row) * K + k0 + cc * 8);
    async_ld16(&W3s[buf][row * 32 + cc * 8], W3 + (size_t)(n0 + row) * K + k0 + cc * 8);
  };

  stage(0, 0);
  const int nk = K >> 5;
  for (int ki = 0; ki < nk; ki++) {
    const int cur = ki & 1;
    wait_async0();
    __syncthreads();
    if (ki + 1 < nk) stage(cur ^ 1, (ki + 1) * 32);
    v16bf a = frag_a(&As[cur][wave * 16 * 32], 32, lane, 0);
    v16bf b1f[4], b3f[4];
#pragma unroll
    for (int nt = 0; nt < 4; nt++) {
      b1f[nt] = frag_b(&W1s[cur][nt * 16 * 32], 32, lane, 0);
      b3f[nt] = frag_b(&W3s[cur][nt * 16 * 32], 32, lane, 0);
    }
#pragma unroll
    for (int nt = 0; nt < 4; nt++) {
      acc1[nt] = wmma_bf16(a, b1f[nt], acc1[nt]);
      acc3[nt] = wmma_bf16(a, b3f[nt], acc3[nt]);
    }
    __syncthreads();
  }

#pragma unroll
  for (int nt = 0; nt < 4; nt++)
#pragma unroll
    for (int r = 0; r < 8; r++) {
      int mm = m0 + wave * 16 + r + 8 * half;
      int nn = n0 + nt * 16 + lane15;
      float v1 = acc1[nt][r], v3 = acc3[nt][r];
      float sig = 1.0f / (1.0f + __expf(-v1));
      G[(size_t)mm * N + nn] = f32_bf16(v1 * sig * v3);
    }
}

// ---------------- RoPE on Q and K (bf16 in-place), layout [b,h,s,dh] ----------------
__global__ void rope_kernel(unsigned short* __restrict__ q, unsigned short* __restrict__ k) {
  size_t t = (size_t)blockIdx.x * 256 + threadIdx.x;  // over B*H*S*(DH/2)
  int i = (int)(t & 63);
  size_t row = t >> 6;
  int s = (int)(row & (S_ - 1));
  float ang = (float)s * __powf(10000.0f, -(float)i * (1.0f / 64.0f));
  float sn, cs;
  __sincosf(ang, &sn, &cs);
  size_t base = row * DH_ + 2 * i;
  {
    float a = bf16_f32(q[base]), b = bf16_f32(q[base + 1]);
    q[base] = f32_bf16(a * cs - b * sn);
    q[base + 1] = f32_bf16(a * sn + b * cs);
  }
  {
    float a = bf16_f32(k[base]), b = bf16_f32(k[base + 1]);
    k[base] = f32_bf16(a * cs - b * sn);
    k[base + 1] = f32_bf16(a * sn + b * cs);
  }
}

// ---------------- flash attention (causal), bf16 WMMA, fp32 online softmax ----------------
// Q,K: [b,h,s,dh] bf16.  Vt: [b,h,dh,s] bf16.  ctx out: [b,s,D] bf16.
__global__ __launch_bounds__(256, 1) void attn_kernel(const unsigned short* __restrict__ Q,
                                                      const unsigned short* __restrict__ Km,
                                                      const unsigned short* __restrict__ Vt,
                                                      unsigned short* __restrict__ ctx) {
  __shared__ __align__(16) unsigned short Plds[8 * 16 * 72];
  const int tid = threadIdx.x;
  const int wave = tid >> 5, lane = tid & 31;
  const int lane15 = lane & 15, half = lane >> 4;
  const int qblk = blockIdx.x, bh = blockIdx.y;
  const int q0 = qblk * 128 + wave * 16;
  const float scale = 0.08838834764831845f;  // 1/sqrt(128)

  const unsigned short* qbase = Q + ((size_t)bh * S_ + q0) * DH_;
  v16bf qf[4];
#pragma unroll
  for (int ks = 0; ks < 4; ks++) qf[ks] = frag_a(qbase, DH_, lane, ks * 32);

  v8f acco[8];
  float mrow[8], lrow[8];
#pragma unroll
  for (int dt = 0; dt < 8; dt++)
#pragma unroll
    for (int i = 0; i < 8; i++) acco[dt][i] = 0.f;
#pragma unroll
  for (int r = 0; r < 8; r++) { mrow[r] = -INFINITY; lrow[r] = 0.f; }

  unsigned short* pl = Plds + wave * 16 * 72;
  const int nkt = 2 * (qblk + 1);  // key tiles of 64, uniform across the block
  for (int kt = 0; kt < nkt; ++kt) {
    const int key0 = kt * 64;
    v8f sc[4];
#pragma unroll
    for (int nt = 0; nt < 4; nt++)
#pragma unroll
      for (int i = 0; i < 8; i++) sc[nt][i] = 0.f;

#pragma unroll
    for (int nt = 0; nt < 4; nt++) {
      const unsigned short* kb = Km + ((size_t)bh * S_ + key0 + nt * 16) * DH_;
      v16bf kf[4];
#pragma unroll
      for (int ks = 0; ks < 4; ks++) kf[ks] = frag_b(kb, DH_, lane, ks * 32);
#pragma unroll
      for (int ks = 0; ks < 4; ks++) sc[nt] = wmma_bf16(qf[ks], kf[ks], sc[nt]);
    }

    float mnew[8];
#pragma unroll
    for (int r = 0; r < 8; r++) mnew[r] = -INFINITY;
#pragma unroll
    for (int nt = 0; nt < 4; nt++) {
      int key = key0 + nt * 16 + lane15;
#pragma unroll
      for (int r = 0; r < 8; r++) {
        int qq = q0 + r + 8 * half;
        float s = sc[nt][r] * scale;
        s = (key <= qq) ? s : -INFINITY;
        sc[nt][r] = s;
        mnew[r] = fmaxf(mnew[r], s);
      }
    }
#pragma unroll
    for (int r = 0; r < 8; r++) {
      float v = mnew[r];
      v = fmaxf(v, __shfl_xor(v, 1, 16));
      v = fmaxf(v, __shfl_xor(v, 2, 16));
      v = fmaxf(v, __shfl_xor(v, 4, 16));
      v = fmaxf(v, __shfl_xor(v, 8, 16));
      float mi = fmaxf(mrow[r], v);
      float al = __expf(mrow[r] - mi);
      mrow[r] = mi;
      lrow[r] *= al;
#pragma unroll
      for (int dt = 0; dt < 8; dt++) acco[dt][r] *= al;
    }
    float rs[8];
#pragma unroll
    for (int r = 0; r < 8; r++) rs[r] = 0.f;
#pragma unroll
    for (int nt = 0; nt < 4; nt++)
#pragma unroll
      for (int r = 0; r < 8; r++) {
        float p = __expf(sc[nt][r] - mrow[r]);
        sc[nt][r] = p;
        rs[r] += p;
      }
#pragma unroll
    for (int r = 0; r < 8; r++) {
      float v = rs[r];
      v += __shfl_xor(v, 1, 16);
      v += __shfl_xor(v, 2, 16);
      v += __shfl_xor(v, 4, 16);
      v += __shfl_xor(v, 8, 16);
      lrow[r] += v;
    }

    // C-layout -> A-layout via per-wave LDS buffer (wave-ordered DS ops, no barrier needed)
#pragma unroll
    for (int nt = 0; nt < 4; nt++)
#pragma unroll
      for (int r = 0; r < 8; r++)
        pl[(size_t)(r + 8 * half) * 72 + nt * 16 + lane15] = f32_bf16(sc[nt][r]);

    v16bf pf0 = frag_a(pl, 72, lane, 0);
    v16bf pf1 = frag_a(pl, 72, lane, 32);
#pragma unroll
    for (int dt = 0; dt < 8; dt++) {
      const unsigned short* vb = Vt + ((size_t)bh * DH_ + dt * 16) * S_ + key0;
      v16bf v0 = frag_b(vb, S_, lane, 0);
      v16bf v1 = frag_b(vb, S_, lane, 32);
      acco[dt] = wmma_bf16(pf0, v0, acco[dt]);
      acco[dt] = wmma_bf16(pf1, v1, acco[dt]);
    }
  }

  const int bb = bh / H_, hh = bh % H_;
#pragma unroll
  for (int r = 0; r < 8; r++) {
    float inv = 1.0f / lrow[r];
    int qq = q0 + r + 8 * half;
    unsigned short* orow = ctx + ((size_t)(bb * S_ + qq)) * D_ + hh * DH_;
#pragma unroll
    for (int dt = 0; dt < 8; dt++) orow[dt * 16 + lane15] = f32_bf16(acco[dt][r] * inv);
  }
}

// ---------------- host launcher ----------------
extern "C" void kernel_launch(void* const* d_in, const int* in_sizes, int n_in,
                              void* d_out, int out_size, void* d_ws, size_t ws_size,
                              hipStream_t stream) {
  (void)in_sizes; (void)n_in; (void)out_size; (void)ws_size;
  const float* x   = (const float*)d_in[0];
  const float* wq  = (const float*)d_in[1];
  const float* wk  = (const float*)d_in[2];
  const float* wv  = (const float*)d_in[3];
  const float* wo  = (const float*)d_in[4];
  const float* ln1 = (const float*)d_in[5];
  const float* ln2 = (const float*)d_in[6];
  const float* w1  = (const float*)d_in[7];
  const float* w2  = (const float*)d_in[8];
  const float* w3  = (const float*)d_in[9];
  float* out = (float*)d_out;

  char* base = (char*)d_ws;
  size_t off = 0;
  auto take = [&](size_t bytes) -> void* {
    void* r = base + off;
    off += (bytes + 255) & ~(size_t)255;
    return r;
  };
  unsigned short* hB   = (unsigned short*)take((size_t)BS_ * D_ * 2);
  unsigned short* qB   = (unsigned short*)take((size_t)BS_ * D_ * 2);
  unsigned short* kB   = (unsigned short*)take((size_t)BS_ * D_ * 2);
  unsigned short* vtB  = (unsigned short*)take((size_t)BS_ * D_ * 2);
  unsigned short* ctxB = (unsigned short*)take((size_t)BS_ * D_ * 2);
  float*          yF   = (float*)take((size_t)BS_ * D_ * 4);
  unsigned short* h2B  = (unsigned short*)take((size_t)BS_ * D_ * 2);
  unsigned short* gB   = (unsigned short*)take((size_t)BS_ * DFF_ * 2);
  unsigned short* wqB  = (unsigned short*)take((size_t)D_ * D_ * 2);
  unsigned short* wkB  = (unsigned short*)take((size_t)D_ * D_ * 2);
  unsigned short* wvB  = (unsigned short*)take((size_t)D_ * D_ * 2);
  unsigned short* woB  = (unsigned short*)take((size_t)D_ * D_ * 2);
  unsigned short* w1B  = (unsigned short*)take((size_t)DFF_ * D_ * 2);
  unsigned short* w3B  = (unsigned short*)take((size_t)DFF_ * D_ * 2);
  unsigned short* w2B  = (unsigned short*)take((size_t)D_ * DFF_ * 2);

  auto cvt = [&](const float* in, unsigned short* o, size_t n) {
    cvt_bf16_kernel<<<dim3(2048), dim3(256), 0, stream>>>(in, o, n);
  };
  cvt(wq, wqB, (size_t)D_ * D_);
  cvt(wk, wkB, (size_t)D_ * D_);
  cvt(wv, wvB, (size_t)D_ * D_);
  cvt(wo, woB, (size_t)D_ * D_);
  cvt(w1, w1B, (size_t)DFF_ * D_);
  cvt(w3, w3B, (size_t)DFF_ * D_);
  cvt(w2, w2B, (size_t)D_ * DFF_);

  rmsnorm_kernel<<<dim3(BS_), dim3(256), 0, stream>>>(x, ln1, hB);

  dim3 gDD(D_ / 64, BS_ / 128);
  gemm_ws_kernel<1><<<gDD, 256, 0, stream>>>(hB, wqB, nullptr, nullptr, qB, BS_, D_, D_);
  gemm_ws_kernel<1><<<gDD, 256, 0, stream>>>(hB, wkB, nullptr, nullptr, kB, BS_, D_, D_);
  gemm_ws_kernel<2><<<gDD, 256, 0, stream>>>(hB, wvB, nullptr, nullptr, vtB, BS_, D_, D_);

  rope_kernel<<<dim3((B_ * H_ * S_ * (DH_ / 2)) / 256), dim3(256), 0, stream>>>(qB, kB);

  attn_kernel<<<dim3(S_ / 128, B_ * H_), dim3(256), 0, stream>>>(qB, kB, vtB, ctxB);

  gemm_ws_kernel<0><<<gDD, 256, 0, stream>>>(ctxB, woB, x, yF, nullptr, BS_, D_, D_);

  rmsnorm_kernel<<<dim3(BS_), dim3(256), 0, stream>>>(yF, ln2, h2B);

  ffn_gate_kernel<<<dim3(DFF_ / 64, BS_ / 128), dim3(256), 0, stream>>>(h2B, w1B, w3B, gB, BS_, DFF_, D_);

  gemm_ws_kernel<0><<<gDD, 256, 0, stream>>>(gB, w2B, yF, out, nullptr, BS_, D_, DFF_);
}
